// GATResidualBlock_90580860273089
// MI455X (gfx1250) — compile-verified, matrix operand order checked by reading, not run on hardware
//
#include <hip/hip_runtime.h>
#include <hip/hip_bf16.h>

// ---------------- problem constants (match reference) ----------------
#define NN    100000      // nodes
#define INC   128         // in channels
#define OUTC  256         // out channels
#define HH    4           // heads
#define CC    64          // per-head channels
#define EDIMC 16          // edge attr dim
#define EE    400000      // edges
#define NCAT  768         // W_l | W_r | W_res concatenated columns
#define ETOT  (EE + NN)   // edges incl. self loops
#define SLOPE 0.2f
#define BN_EPS 1e-5f

typedef _Float16 half_t;
typedef __attribute__((ext_vector_type(16))) _Float16 v16h;
typedef __attribute__((ext_vector_type(8)))  float    v8f;

union FragU { v16h h; uint4 u[2]; };

// ---------------- utility fills ----------------
__global__ void fill_f32_kernel(float* __restrict__ p, float v, size_t n) {
  size_t i = (size_t)blockIdx.x * blockDim.x + threadIdx.x;
  if (i < n) p[i] = v;
}

// ---------------- convert x to f16 ----------------
__global__ void cvt_x_kernel(const float* __restrict__ x, half_t* __restrict__ xh, size_t n) {
  size_t i = (size_t)blockIdx.x * blockDim.x + threadIdx.x;
  if (i < n) xh[i] = (half_t)x[i];
}

// ---------------- pack concatenated weights into WMMA-B fragment layout ----------------
// Fragment layout for V_WMMA_F32_16X16X32_F16 B operand (32x16, f16):
//   lane = n + 16*lanehalf,  lanehalf = (kk>>3)&1,  elem i = (kk&7) + 8*((kk>>4)&1)
// Storage: [ct(48)][kt(4)][lane(32)][elem(16)] halves -> each lane reads 32 contiguous bytes.
__global__ void pack_w_kernel(const float* __restrict__ Wl, const float* __restrict__ Wr,
                              const float* __restrict__ Wres,
                              const float* __restrict__ bl, const float* __restrict__ br,
                              half_t* __restrict__ wp, float* __restrict__ biasc) {
  int idx = blockIdx.x * blockDim.x + threadIdx.x;   // over 128*768
  if (idx >= INC * NCAT) return;
  int k = idx / NCAT;
  int c = idx - k * NCAT;
  float v;
  if (c < 256)      v = Wl[k * 256 + c];
  else if (c < 512) v = Wr[k * 256 + (c - 256)];
  else              v = Wres[k * 256 + (c - 512)];
  int ct = c >> 4, n = c & 15;
  int kt = k >> 5, kk = k & 31;
  int lanehalf = (kk >> 3) & 1;
  int elem = (kk & 7) + 8 * ((kk >> 4) & 1);
  wp[(((size_t)(ct * 4 + kt) * 32) + (n + 16 * lanehalf)) * 16 + elem] = (half_t)v;
  if (k == 0) {
    biasc[c] = (c < 256) ? bl[c] : (c < 512 ? br[c - 256] : 0.0f);
  }
}

// ---------------- self-loop attr: segment mean of incoming edge attrs ----------------
__global__ void seg_edge_attr_kernel(const long long* __restrict__ ei,
                                     const float* __restrict__ eattr,
                                     float* __restrict__ sum_attr, float* __restrict__ cnt) {
  size_t idx = (size_t)blockIdx.x * blockDim.x + threadIdx.x;   // over E*16
  if (idx >= (size_t)EE * EDIMC) return;
  size_t e = idx >> 4;
  int k = (int)(idx & 15);
  long long d = ei[EE + e];
  atomicAdd(&sum_attr[(size_t)d * EDIMC + k], eattr[idx]);
  if (k == 0) atomicAdd(&cnt[d], 1.0f);
}

__global__ void loop_div_kernel(float* __restrict__ sum_attr, const float* __restrict__ cnt) {
  size_t idx = (size_t)blockIdx.x * blockDim.x + threadIdx.x;   // over N*16
  if (idx >= (size_t)NN * EDIMC) return;
  size_t r = idx >> 4;
  sum_attr[idx] = sum_attr[idx] / fmaxf(cnt[r], 1.0f);
}

// ---------------- fused WMMA GEMM: XCAT[N][768] = x @ [W_l|W_r|W_res] + [b_l|b_r|0] ----------------
__device__ inline v16h load_a_frag_lds(const half_t* __restrict__ As, int kt, int lane) {
  int m = lane & 15;
  int k0 = (lane >> 4) << 3;
  FragU f;
  f.u[0] = *(const uint4*)(&As[m * INC + kt + k0]);        // kk k0..k0+7
  f.u[1] = *(const uint4*)(&As[m * INC + kt + 16 + k0]);   // kk 16+k0..16+k0+7
  return f.h;
}

__global__ __launch_bounds__(256) void gemm_wmma_kernel(const half_t* __restrict__ xh,
                                                        const half_t* __restrict__ wp,
                                                        const float* __restrict__ biasc,
                                                        float* __restrict__ xcat) {
  __shared__ half_t As[16 * INC];
  const int rt = blockIdx.x * 16;
  // cooperative A-tile load: 16 rows x 128 k, 256 threads x 8 halves (16B) each
  {
    int t = threadIdx.x * 8;
    *(uint4*)(&As[t]) = *(const uint4*)(&xh[(size_t)rt * INC + t]);
  }
  __syncthreads();

  const int wave = threadIdx.x >> 5;
  const int lane = threadIdx.x & 31;

  v16h a[4];
#pragma unroll
  for (int kt = 0; kt < 4; ++kt) a[kt] = load_a_frag_lds(As, kt * 32, lane);

#pragma unroll
  for (int j = 0; j < 6; ++j) {
    const int col = wave * 16 + j * 128;   // 8 waves x 6 tiles cover 768 cols
    const int ct = col >> 4;
    v8f acc = {};
#pragma unroll
    for (int kt = 0; kt < 4; ++kt) {
      FragU b;
      const half_t* bp = &wp[(((size_t)(ct * 4 + kt) * 32) + lane) * 16];
      b.u[0] = *(const uint4*)(bp);
      b.u[1] = *(const uint4*)(bp + 8);
      acc = __builtin_amdgcn_wmma_f32_16x16x32_f16(false, a[kt], false, b.h,
                                                   (short)0, acc, false, false);
    }
    const int n  = lane & 15;
    const int mo = (lane >> 4) << 3;
    const float bv = biasc[col + n];
#pragma unroll
    for (int v = 0; v < 8; ++v)
      xcat[(size_t)(rt + v + mo) * NCAT + col + n] = acc[v] + bv;
  }
}

// ---------------- float atomic max via signed/unsigned trick ----------------
__device__ inline void atomicMaxF(float* addr, float val) {
  if (val >= 0.0f) atomicMax((int*)addr, __float_as_int(val));
  else             atomicMin((unsigned int*)addr, __float_as_uint(val));
}

// ---------------- pass 1: attention scores e[E',H] + segment max ----------------
__global__ __launch_bounds__(256) void edge_score_kernel(const long long* __restrict__ ei,
                                                         const float* __restrict__ eattr,
                                                         const float* __restrict__ loop_attr,
                                                         const float* __restrict__ xcat,
                                                         const float* __restrict__ We,
                                                         const float* __restrict__ att,
                                                         float* __restrict__ escore,
                                                         float* __restrict__ emax) {
  const int wid = blockIdx.x * 8 + (threadIdx.x >> 5);   // wave per edge
  if (wid >= ETOT) return;
  const int lane = threadIdx.x & 31;

  long long s, d;
  const float* ea;
  if (wid < EE) { s = ei[wid]; d = ei[EE + wid]; ea = eattr + (size_t)wid * EDIMC; }
  else          { s = d = wid - EE;              ea = loop_attr + (size_t)(wid - EE) * EDIMC; }

  const float eav = (lane < EDIMC) ? ea[lane] : 0.0f;
  const int cbase = lane * 8;
  const float* xlp = xcat + (size_t)s * NCAT + cbase;          // xl = cols 0..255
  const float* xrp = xcat + (size_t)d * NCAT + 256 + cbase;    // xr = cols 256..511

  float z[8];
#pragma unroll
  for (int j = 0; j < 8; ++j) z[j] = xlp[j] + xrp[j];
#pragma unroll
  for (int k = 0; k < EDIMC; ++k) {
    const float a = __shfl(eav, k, 32);
    const float* wrow = We + k * OUTC + cbase;
#pragma unroll
    for (int j = 0; j < 8; ++j) z[j] = fmaf(a, wrow[j], z[j]);
  }
  float part = 0.0f;
#pragma unroll
  for (int j = 0; j < 8; ++j) {
    const float t = z[j] > 0.0f ? z[j] : SLOPE * z[j];
    part = fmaf(t, att[cbase + j], part);                      // att flat index == channel
  }
  part += __shfl_xor(part, 1, 32);
  part += __shfl_xor(part, 2, 32);
  part += __shfl_xor(part, 4, 32);
  if ((lane & 7) == 0) {
    const int h = lane >> 3;
    escore[(size_t)wid * HH + h] = part;
    atomicMaxF(&emax[(size_t)d * HH + h], part);
  }
}

// ---------------- pass 2: exp(e - max) and segment denom ----------------
__global__ void edge_exp_kernel(const long long* __restrict__ ei,
                                float* __restrict__ escore,
                                const float* __restrict__ emax,
                                float* __restrict__ denom) {
  size_t idx = (size_t)blockIdx.x * blockDim.x + threadIdx.x;   // over ETOT*H
  if (idx >= (size_t)ETOT * HH) return;
  size_t e = idx >> 2;
  int h = (int)(idx & 3);
  long long d = (e < EE) ? ei[EE + e] : (long long)(e - EE);
  float ex = __expf(escore[idx] - emax[(size_t)d * HH + h]);
  escore[idx] = ex;
  atomicAdd(&denom[(size_t)d * HH + h], ex);
}

// ---------------- pass 3: scatter alpha * xl[src] ----------------
__global__ __launch_bounds__(256) void edge_aggregate_kernel(const long long* __restrict__ ei,
                                                             const float* __restrict__ xcat,
                                                             const float* __restrict__ escore,
                                                             const float* __restrict__ denom,
                                                             float* __restrict__ out_acc) {
  const int wid = blockIdx.x * 8 + (threadIdx.x >> 5);
  if (wid >= ETOT) return;
  const int lane = threadIdx.x & 31;
  long long s, d;
  if (wid < EE) { s = ei[wid]; d = ei[EE + wid]; }
  else          { s = d = wid - EE; }
  const int h = lane >> 3;
  const float alpha = escore[(size_t)wid * HH + h] / denom[(size_t)d * HH + h];
  const int cbase = lane * 8;
  const float* xlp = xcat + (size_t)s * NCAT + cbase;
  float* op = out_acc + (size_t)d * OUTC + cbase;
#pragma unroll
  for (int j = 0; j < 8; ++j) atomicAdd(&op[j], alpha * xlp[j]);
}

// ---------------- BN batch statistics ----------------
__global__ __launch_bounds__(256) void bn_stats_kernel(const float* __restrict__ out_acc,
                                                       float* __restrict__ bn_sum,
                                                       float* __restrict__ bn_sq) {
  const int c = threadIdx.x;          // 256 channels
  float s = 0.0f, q = 0.0f;
  for (int r = blockIdx.x; r < NN; r += gridDim.x) {
    const float v = out_acc[(size_t)r * OUTC + c];
    s += v; q += v * v;
  }
  atomicAdd(&bn_sum[c], s);
  atomicAdd(&bn_sq[c], q);
}

// ---------------- finalize: BN + ELU + residual (xres = XCAT cols 512..767) ----------------
__global__ void finalize_kernel(float* __restrict__ out,
                                const float* __restrict__ xcat,
                                const float* __restrict__ bn_sum,
                                const float* __restrict__ bn_sq,
                                const float* __restrict__ gamma,
                                const float* __restrict__ beta) {
  size_t idx = (size_t)blockIdx.x * blockDim.x + threadIdx.x;   // over N*256
  if (idx >= (size_t)NN * OUTC) return;
  size_t r = idx >> 8;
  int c = (int)(idx & 255);
  const float invN = 1.0f / (float)NN;
  const float mean = bn_sum[c] * invN;
  const float var  = bn_sq[c] * invN - mean * mean;
  const float v = out[idx];
  float nrm = (v - mean) * rsqrtf(var + BN_EPS) * gamma[c] + beta[c];
  float el = nrm > 0.0f ? nrm : (__expf(nrm) - 1.0f);
  out[idx] = el + xcat[r * NCAT + 512 + c];
}

// ---------------- host orchestration ----------------
static inline size_t alignup(size_t x) { return (x + 255) & ~(size_t)255; }

extern "C" void kernel_launch(void* const* d_in, const int* in_sizes, int n_in,
                              void* d_out, int out_size, void* d_ws, size_t ws_size,
                              hipStream_t stream) {
  const float*      x     = (const float*)d_in[0];
  const long long*  ei    = (const long long*)d_in[1];
  const float*      eattr = (const float*)d_in[2];
  const float*      Wl    = (const float*)d_in[3];
  const float*      bl    = (const float*)d_in[4];
  const float*      Wr    = (const float*)d_in[5];
  const float*      br    = (const float*)d_in[6];
  const float*      We    = (const float*)d_in[7];
  const float*      att   = (const float*)d_in[8];
  // d_in[9] = bias: cancels under training-mode BatchNorm (mean shift), skipped
  const float*      gamma = (const float*)d_in[10];
  const float*      beta  = (const float*)d_in[11];
  const float*      Wres  = (const float*)d_in[12];
  float* out = (float*)d_out;   // also serves as the scatter accumulator

  // workspace carve-up
  char* w = (char*)d_ws;
  size_t off = 0;
  half_t* xh      = (half_t*)(w + off); off += alignup((size_t)NN * INC * sizeof(half_t));
  half_t* wp      = (half_t*)(w + off); off += alignup((size_t)INC * NCAT * sizeof(half_t));
  float* biasc    = (float*)(w + off);  off += alignup((size_t)NCAT * sizeof(float));
  float* xcat     = (float*)(w + off);  off += alignup((size_t)NN * NCAT * sizeof(float));
  float* sum_attr = (float*)(w + off);  off += alignup((size_t)NN * EDIMC * sizeof(float));
  float* cnt      = (float*)(w + off);  off += alignup((size_t)NN * sizeof(float));
  float* escore   = (float*)(w + off);  off += alignup((size_t)ETOT * HH * sizeof(float));
  float* emax     = (float*)(w + off);  off += alignup((size_t)NN * HH * sizeof(float));
  float* denom    = (float*)(w + off);  off += alignup((size_t)NN * HH * sizeof(float));
  float* bn_sum   = (float*)(w + off);  off += alignup(256 * sizeof(float));
  float* bn_sq    = (float*)(w + off);  off += alignup(256 * sizeof(float));
  (void)ws_size; (void)n_in; (void)in_sizes; (void)out_size;

  const int T = 256;
  auto blocks = [](size_t n, int t) { return (unsigned)((n + t - 1) / t); };

  // 0) zero / init accumulators (every call: scatter targets must start clean)
  fill_f32_kernel<<<blocks((size_t)NN * OUTC, T), T, 0, stream>>>(out, 0.0f, (size_t)NN * OUTC);
  fill_f32_kernel<<<blocks((size_t)NN * EDIMC, T), T, 0, stream>>>(sum_attr, 0.0f, (size_t)NN * EDIMC);
  fill_f32_kernel<<<blocks((size_t)NN, T), T, 0, stream>>>(cnt, 0.0f, (size_t)NN);
  fill_f32_kernel<<<blocks((size_t)NN * HH, T), T, 0, stream>>>(emax, -1e30f, (size_t)NN * HH);
  fill_f32_kernel<<<blocks((size_t)NN * HH, T), T, 0, stream>>>(denom, 0.0f, (size_t)NN * HH);
  fill_f32_kernel<<<1, T, 0, stream>>>(bn_sum, 0.0f, 256);
  fill_f32_kernel<<<1, T, 0, stream>>>(bn_sq, 0.0f, 256);

  // 1) precision conversion + weight packing for WMMA fragments
  cvt_x_kernel<<<blocks((size_t)NN * INC, T), T, 0, stream>>>(x, xh, (size_t)NN * INC);
  pack_w_kernel<<<blocks((size_t)INC * NCAT, T), T, 0, stream>>>(Wl, Wr, Wres, bl, br, wp, biasc);

  // 2) self-loop edge attr = mean of incoming edge attrs
  seg_edge_attr_kernel<<<blocks((size_t)EE * EDIMC, T), T, 0, stream>>>(ei, eattr, sum_attr, cnt);
  loop_div_kernel<<<blocks((size_t)NN * EDIMC, T), T, 0, stream>>>(sum_attr, cnt);

  // 3) fused WMMA GEMM: XCAT = x @ [W_l|W_r|W_res] + [b_l|b_r|0]
  gemm_wmma_kernel<<<NN / 16, T, 0, stream>>>(xh, wp, biasc, xcat);

  // 4) attention + segment softmax + aggregation
  edge_score_kernel<<<ETOT / 8, T, 0, stream>>>(ei, eattr, sum_attr, xcat, We, att, escore, emax);
  edge_exp_kernel<<<blocks((size_t)ETOT * HH, T), T, 0, stream>>>(ei, escore, emax, denom);
  edge_aggregate_kernel<<<ETOT / 8, T, 0, stream>>>(ei, xcat, escore, denom, out);

  // 5) BatchNorm stats + finalize (BN -> ELU -> +residual)
  bn_stats_kernel<<<512, T, 0, stream>>>(out, bn_sum, bn_sq);
  finalize_kernel<<<blocks((size_t)NN * OUTC, T), T, 0, stream>>>(out, xcat, bn_sum, bn_sq, gamma, beta);
}
